// LocalBlock_18588618457411
// MI455X (gfx1250) — matrix operand order repaired and checked by reading, not compiled, not run on hardware
//
#include <hip/hip_runtime.h>

// ---------------- types ----------------
typedef __attribute__((ext_vector_type(16))) __bf16 v16bf;
typedef __attribute__((ext_vector_type(8)))  float  v8f;

union FragBF { uint4 q[2]; v16bf v; };

__device__ __forceinline__ unsigned short f2bf(float f) {
    unsigned int u = __builtin_bit_cast(unsigned int, f);
    unsigned int r = (u + 0x7FFFu + ((u >> 16) & 1u)) >> 16;
    return (unsigned short)r;
}
__device__ __forceinline__ float bf2f(unsigned short h) {
    unsigned int u = ((unsigned int)h) << 16;
    return __builtin_bit_cast(float, u);
}

#define MTOK 25088   // B*H*W
#define HH 56
#define WW 56
#define HWP 3136     // 56*56
#define CC 384
#define TILE_K 32
#define BM 128
#define BN 128
#define NBUF 4
#define ABUF_BYTES (BM * TILE_K * 2)
#define BBUF_BYTES (BN * TILE_K * 2)

// async copy 32B (two b128) global -> LDS; INST_OFFSET applies to both addresses
#define ASYNC_CP32(ldsoff, gptr)                                              \
    asm volatile("global_load_async_to_lds_b128 %0, %1, off\n\t"              \
                 "global_load_async_to_lds_b128 %0, %1, off offset:16"        \
                 :: "v"(ldsoff),                                              \
                    "v"((unsigned long long)(uintptr_t)(gptr))                \
                 : "memory")

#define WAITA(n) asm volatile("s_wait_asynccnt " #n ::: "memory")

// ---------------- f32 -> bf16 convert ----------------
__global__ __launch_bounds__(256) void f32_to_bf16_kernel(
    const float* __restrict__ in, unsigned short* __restrict__ out, int n) {
    int i = blockIdx.x * 256 + threadIdx.x;
    if (i < n) out[i] = f2bf(in[i]);
}

// ---------------- LN over C with NCHW->NHWC transpose (bf16 out) ----------------
// thread-per-pixel: channel-loop reads are coalesced across threads (NCHW)
__global__ __launch_bounds__(256) void ln_transpose_kernel(
    const float* __restrict__ x, const float* __restrict__ g,
    const float* __restrict__ bt, unsigned short* __restrict__ out) {
    int m = blockIdx.x * 256 + threadIdx.x;
    if (m >= MTOK) return;
    int bb = m / HWP, rem = m % HWP;
    const float* base = x + (size_t)bb * CC * HWP + rem;   // + c*HWP
    float s = 0.f, s2 = 0.f;
    for (int c = 0; c < CC; ++c) { float v = base[(size_t)c * HWP]; s += v; s2 += v * v; }
    float mean = s * (1.0f / CC);
    float var  = s2 * (1.0f / CC) - mean * mean;
    float rstd = rsqrtf(var + 1e-5f);
    unsigned short* op = out + (size_t)m * CC;
    for (int c = 0; c < CC; ++c) {
        float v = (base[(size_t)c * HWP] - mean) * rstd * g[c] + bt[c];
        op[c] = f2bf(v);
    }
}

// ---------------- LN over contiguous rows (f32 in, bf16 out) ----------------
// wave-per-pixel: lane-strided channels -> coalesced reads and writes
__global__ __launch_bounds__(256) void ln_rows_kernel(
    const float* __restrict__ in, const float* __restrict__ g,
    const float* __restrict__ bt, unsigned short* __restrict__ out) {
    int m    = blockIdx.x * 8 + (threadIdx.x >> 5);
    int lane = threadIdx.x & 31;
    if (m >= MTOK) return;
    const float* base = in + (size_t)m * CC;
    float vals[CC / 32];
    float s = 0.f, s2 = 0.f;
#pragma unroll
    for (int i = 0; i < CC / 32; ++i) {
        float v = base[lane + 32 * i];
        vals[i] = v; s += v; s2 += v * v;
    }
#pragma unroll
    for (int off = 16; off > 0; off >>= 1) {
        s  += __shfl_xor(s, off, 32);
        s2 += __shfl_xor(s2, off, 32);
    }
    float mean = s * (1.0f / CC);
    float var  = s2 * (1.0f / CC) - mean * mean;
    float rstd = rsqrtf(var + 1e-5f);
    unsigned short* op = out + (size_t)m * CC;
#pragma unroll
    for (int i = 0; i < CC / 32; ++i) {
        int c = lane + 32 * i;
        op[c] = f2bf((vals[i] - mean) * rstd * g[c] + bt[c]);
    }
}

// ---------------- dilated local attention (one wave per pixel*head) ----------
__global__ __launch_bounds__(256) void dilate_attn_kernel(
    const unsigned short* __restrict__ qkv,
    const unsigned short* __restrict__ xn,
    unsigned short* __restrict__ attn_in) {
    int wid  = (blockIdx.x * 256 + threadIdx.x) >> 5;   // global wave id
    int lane = threadIdx.x & 31;                         // head-dim channel (hd=32)
    int head = wid % 12;
    int m    = wid / 12;
    int br   = head >> 2;          // dilation branch 0..2
    int hn   = head & 3;           // head within branch
    int dil  = br + 1;
    int bb = m / HWP, rem = m % HWP, h = rem / WW, w = rem % WW;
    int cbase = br * 128 + hn * 32 + lane;

    const float scale = 0.17677669529663687f;   // 1/sqrt(32)
    float q = bf2f(qkv[(size_t)m * 1152 + cbase]) * scale;

    float logits[9];
#pragma unroll
    for (int kk = 0; kk < 9; ++kk) {
        int di = kk / 3 - 1, dj = kk % 3 - 1;
        int hh = h + di * dil, ww2 = w + dj * dil;
        float partial = 0.f;
        if (hh >= 0 && hh < HH && ww2 >= 0 && ww2 < WW) {
            int mm = bb * HWP + hh * WW + ww2;
            partial = q * bf2f(qkv[(size_t)mm * 1152 + 384 + cbase]);
        }
#pragma unroll
        for (int off = 16; off > 0; off >>= 1)
            partial += __shfl_xor(partial, off, 32);
        logits[kk] = partial;      // zero-padded OOB -> logit 0, matches unfold
    }

    float mx = logits[0];
#pragma unroll
    for (int kk = 1; kk < 9; ++kk) mx = fmaxf(mx, logits[kk]);
    float ex[9], ssum = 0.f;
#pragma unroll
    for (int kk = 0; kk < 9; ++kk) { ex[kk] = __expf(logits[kk] - mx); ssum += ex[kk]; }
    float rs = 1.0f / ssum;

    float o = 0.f;
#pragma unroll
    for (int kk = 0; kk < 9; ++kk) {
        int di = kk / 3 - 1, dj = kk % 3 - 1;
        int hh = h + di * dil, ww2 = w + dj * dil;
        if (hh >= 0 && hh < HH && ww2 >= 0 && ww2 < WW) {
            int mm = bb * HWP + hh * WW + ww2;
            o += (ex[kk] * rs) * bf2f(qkv[(size_t)mm * 1152 + 768 + cbase]);
        }
    }

    float xnf = bf2f(xn[(size_t)m * CC + cbase]);
    attn_in[(size_t)m * CC + cbase] = f2bf(o + xnf);
}

// ---------------- WMMA bf16 GEMM:  D[M,N] = A[M,K] * W[N,K]^T  ----------------
// Block: 256 thr = 8 waves. Tile: 128(M) x 128(N), K-step 32.
// Async copy ring: 4 LDS buffers, issue 2 tiles ahead, ONE barrier per K-step.
//   iter t: issue(t+2) -> buf[(t+2)%4]; s_wait_asynccnt<=8 (tile t landed,
//   in-order completion); barrier; compute(t) from buf[t%4].
//   Window w(t-1) holds compute(t-1) on buf[(t-1)%4] and issue(t+2) on
//   buf[(t+2)%4] -- disjoint (distance 3 mod 4): race-free.
// EPI 0: qkv  -> bf16 store
// EPI 1: proj -> f32 out1 = acc + bias + x[b,c,h,w] (NCHW gather)
// EPI 2: fc1  -> bf16 gelu(acc + bias)
// EPI 3: fc2  -> final[b,c,h,w] = acc + bias + out1[m,c] (NCHW scatter)
template<int EPI>
__global__ __launch_bounds__(256) void gemm_bf16_wmma(
    const unsigned short* __restrict__ A,
    const unsigned short* __restrict__ Wt,
    int M, int N, int K,
    unsigned short* __restrict__ outH,
    float* __restrict__ outF,
    const float* __restrict__ bias,
    const float* __restrict__ residual,
    float* __restrict__ finalOut) {
    __shared__ __align__(16) unsigned short ldsA[NBUF][BM * TILE_K];
    __shared__ __align__(16) unsigned short ldsB[NBUF][BN * TILE_K];

    const int t    = threadIdx.x;
    const int lane = t & 31;
    const int wave = t >> 5;
    const int m0   = blockIdx.y * BM;
    const int n0   = blockIdx.x * BN;

    // loaders: 128 rows x 2 halves (16 elems = 32B per thread per matrix)
    const int r    = t >> 1;
    const int half = t & 1;

    const int fr = lane & 15;              // frag row/col within 16
    const int fs = lane >> 4;              // lane-half select

    const unsigned short* gA = A  + (size_t)(m0 + r) * K + half * 16;
    const unsigned short* gB = Wt + (size_t)(n0 + r) * K + half * 16;
    const unsigned ldsOffByte = (unsigned)(r * TILE_K + half * 16) * 2;
    const unsigned aBase = (unsigned)(uintptr_t)&ldsA[0][0] + ldsOffByte;
    const unsigned bBase = (unsigned)(uintptr_t)&ldsB[0][0] + ldsOffByte;

    auto issue = [&](int tile) {
        const unsigned buf = (unsigned)(tile & (NBUF - 1));
        ASYNC_CP32(aBase + buf * ABUF_BYTES, gA + tile * TILE_K);
        ASYNC_CP32(bBase + buf * BBUF_BYTES, gB + tile * TILE_K);
    };

    v8f accs[8];
#pragma unroll
    for (int g = 0; g < 8; ++g) accs[g] = v8f{};

    const int nT = K / TILE_K;
    issue(0);
    if (nT > 1) issue(1);
    for (int tk = 0; tk < nT; ++tk) {
        const int buf = tk & (NBUF - 1);
        if (tk + 2 < nT) { issue(tk + 2); WAITA(8); }
        else if (tk + 1 < nT) { WAITA(4); }
        else { WAITA(0); }
        __syncthreads();   // single barrier per K-step

        // A fragment (16x32): lane-half fs picks K blocks {8fs..} and {16+8fs..}
        FragBF afr;
        {
            const unsigned short* rowp = &ldsA[buf][(wave * 16 + fr) * TILE_K];
            afr.q[0] = *(const uint4*)(rowp + 8 * fs);
            afr.q[1] = *(const uint4*)(rowp + 16 + 8 * fs);
        }
#pragma unroll
        for (int h8 = 0; h8 < 2; ++h8) {
            FragBF bfr[4];
#pragma unroll
            for (int j = 0; j < 4; ++j) {
                // B fragment (32x16): column = fr, K blocks {8fs..} and {16+8fs..}
                const unsigned short* rowp =
                    &ldsB[buf][((h8 * 4 + j) * 16 + fr) * TILE_K + 8 * fs];
                bfr[j].q[0] = *(const uint4*)(rowp);
                bfr[j].q[1] = *(const uint4*)(rowp + 16);
            }
#pragma unroll
            for (int j = 0; j < 4; ++j) {
                accs[h8 * 4 + j] = __builtin_amdgcn_wmma_f32_16x16x32_bf16(
                    false, afr.v, false, bfr[j].v, (short)0, accs[h8 * 4 + j],
                    false, false);
            }
        }
    }

    // epilogue: C/D layout -> M = v + 8*fs, N = fr
    const int mrow_base = m0 + wave * 16 + fs * 8;
#pragma unroll
    for (int g = 0; g < 8; ++g) {
        const int ncol = n0 + g * 16 + fr;
#pragma unroll
        for (int v = 0; v < 8; ++v) {
            const int mrow = mrow_base + v;
            float val = accs[g][v];
            if constexpr (EPI == 0) {
                outH[(size_t)mrow * N + ncol] = f2bf(val);
            } else if constexpr (EPI == 1) {
                int bb = mrow / HWP, rem = mrow % HWP, hh = rem / WW, ww2 = rem % WW;
                float res = residual[(((size_t)bb * CC + ncol) * HH + hh) * WW + ww2];
                outF[(size_t)mrow * N + ncol] = val + bias[ncol] + res;
            } else if constexpr (EPI == 2) {
                float z = val + bias[ncol];
                float ge = 0.5f * z * (1.0f + erff(z * 0.7071067811865476f));
                outH[(size_t)mrow * N + ncol] = f2bf(ge);
            } else {
                int bb = mrow / HWP, rem = mrow % HWP, hh = rem / WW, ww2 = rem % WW;
                float res = residual[(size_t)mrow * CC + ncol];
                finalOut[(((size_t)bb * CC + ncol) * HH + hh) * WW + ww2] =
                    val + bias[ncol] + res;
            }
        }
    }
}

// ---------------- launch ----------------
extern "C" void kernel_launch(void* const* d_in, const int* in_sizes, int n_in,
                              void* d_out, int out_size, void* d_ws, size_t ws_size,
                              hipStream_t stream) {
    const float* x      = (const float*)d_in[0];
    const float* qkv_w  = (const float*)d_in[1];
    const float* proj_w = (const float*)d_in[2];
    const float* proj_b = (const float*)d_in[3];
    const float* n1_g   = (const float*)d_in[4];
    const float* n1_b   = (const float*)d_in[5];
    const float* n2_g   = (const float*)d_in[6];
    const float* n2_b   = (const float*)d_in[7];
    const float* fc1_w  = (const float*)d_in[8];
    const float* fc1_b  = (const float*)d_in[9];
    const float* fc2_w  = (const float*)d_in[10];
    const float* fc2_b  = (const float*)d_in[11];
    float* out = (float*)d_out;

    const size_t M = MTOK;
    char* ws = (char*)d_ws;
    size_t off = 0;
    auto take = [&](size_t bytes) { size_t o = off; off += (bytes + 255) & ~(size_t)255; return o; };
    unsigned short* xn_bf   = (unsigned short*)(ws + take(M * 384 * 2));
    unsigned short* qkv_bf  = (unsigned short*)(ws + take(M * 1152 * 2));
    unsigned short* ain_bf  = (unsigned short*)(ws + take(M * 384 * 2));
    float*          out1    = (float*)         (ws + take(M * 384 * 4));
    unsigned short* x2_bf   = (unsigned short*)(ws + take(M * 384 * 2));
    unsigned short* h_bf    = (unsigned short*)(ws + take(M * 1536 * 2));
    unsigned short* wqkv_bf = (unsigned short*)(ws + take((size_t)1152 * 384 * 2));
    unsigned short* wproj_bf= (unsigned short*)(ws + take((size_t)384 * 384 * 2));
    unsigned short* wfc1_bf = (unsigned short*)(ws + take((size_t)1536 * 384 * 2));
    unsigned short* wfc2_bf = (unsigned short*)(ws + take((size_t)384 * 1536 * 2));
    (void)ws_size; (void)in_sizes; (void)n_in; (void)out_size;

    // weight converts (f32 -> bf16)
    f32_to_bf16_kernel<<<(1152 * 384 + 255) / 256, 256, 0, stream>>>(qkv_w, wqkv_bf, 1152 * 384);
    f32_to_bf16_kernel<<<(384 * 384 + 255) / 256, 256, 0, stream>>>(proj_w, wproj_bf, 384 * 384);
    f32_to_bf16_kernel<<<(1536 * 384 + 255) / 256, 256, 0, stream>>>(fc1_w, wfc1_bf, 1536 * 384);
    f32_to_bf16_kernel<<<(384 * 1536 + 255) / 256, 256, 0, stream>>>(fc2_w, wfc2_bf, 384 * 1536);

    // LN1 + transpose
    ln_transpose_kernel<<<MTOK / 256, 256, 0, stream>>>(x, n1_g, n1_b, xn_bf);

    // qkv GEMM: [M,1152] = xn[M,384] * qkv_w[1152,384]^T
    gemm_bf16_wmma<0><<<dim3(1152 / BN, MTOK / BM), 256, 0, stream>>>(
        xn_bf, wqkv_bf, MTOK, 1152, 384, qkv_bf, nullptr, nullptr, nullptr, nullptr);

    // dilated attention + residual with xn -> proj input
    dilate_attn_kernel<<<(MTOK * 12) / 8, 256, 0, stream>>>(qkv_bf, xn_bf, ain_bf);

    // proj GEMM + bias + residual(x, NCHW gather) -> out1 (f32)
    gemm_bf16_wmma<1><<<dim3(384 / BN, MTOK / BM), 256, 0, stream>>>(
        ain_bf, wproj_bf, MTOK, 384, 384, nullptr, out1, proj_b, x, nullptr);

    // LN2 (wave-per-pixel, coalesced)
    ln_rows_kernel<<<MTOK / 8, 256, 0, stream>>>(out1, n2_g, n2_b, x2_bf);

    // fc1 GEMM + bias + gelu -> h (bf16)
    gemm_bf16_wmma<2><<<dim3(1536 / BN, MTOK / BM), 256, 0, stream>>>(
        x2_bf, wfc1_bf, MTOK, 1536, 384, h_bf, nullptr, fc1_b, nullptr, nullptr);

    // fc2 GEMM + bias + residual(out1) -> final NCHW scatter to d_out
    gemm_bf16_wmma<3><<<dim3(384 / BN, MTOK / BM), 256, 0, stream>>>(
        h_bf, wfc2_bf, MTOK, 384, 1536, nullptr, nullptr, fc2_b, out1, out);
}
